// PositionWiseFFN_Lora_17952963297411
// MI455X (gfx1250) — matrix operand order repaired and checked by reading, not compiled, not run on hardware
//
#include <hip/hip_runtime.h>
#include <hip/hip_bf16.h>

typedef __attribute__((ext_vector_type(16))) __bf16 v16bf;
typedef __attribute__((ext_vector_type(8)))  __bf16 v8bf;
typedef __attribute__((ext_vector_type(4)))  __bf16 v4bf;
typedef __attribute__((ext_vector_type(8)))  float  v8f;

namespace {
constexpr int kB = 4;
constexpr int kS = 2048;
constexpr int kD = 1024;
constexpr int kH = 4096;
constexpr int kM = kB * kS;   // 8192 token rows
constexpr int kLogS = 11;     // S == 2048

constexpr int kLdsStride = 40;                 // bf16 units per row (80 B, conflict-free)
constexpr int kTileA = 256 * kLdsStride;       // A tile: 256 rows
constexpr int kTileB = 128 * kLdsStride;       // B tile: 128 rows

// workspace layout (bytes)
constexpr size_t OFF_W1BF = 0;                                // H*D bf16 =  8 MiB
constexpr size_t OFF_W2BF = OFF_W1BF + (size_t)kH * kD * 2;   // D*H bf16 =  8 MiB
constexpr size_t OFF_XBF  = OFF_W2BF + (size_t)kD * kH * 2;   // M*D bf16 = 16 MiB
constexpr size_t OFF_HBF  = OFF_XBF  + (size_t)kM * kD * 2;   // M*H bf16 = 64 MiB
constexpr size_t OFF_T1   = OFF_HBF  + (size_t)kM * kH * 2;   // M*4  f32
constexpr size_t OFF_T2   = OFF_T1   + (size_t)kM * 4 * 4;    // M*16 f32
} // namespace

// ---------------------------------------------------------------------------
// fp32 -> bf16 conversion (vectorized 4-wide)
// ---------------------------------------------------------------------------
__global__ __launch_bounds__(256) void cvt_f32_to_bf16_x4(
    const float4* __restrict__ src, v4bf* __restrict__ dst, int n4) {
  int i = blockIdx.x * blockDim.x + threadIdx.x;
  const int stride = gridDim.x * blockDim.x;
  for (; i < n4; i += stride) {
    const float4 v = src[i];
    v4bf o;
    o[0] = (__bf16)v.x; o[1] = (__bf16)v.y; o[2] = (__bf16)v.z; o[3] = (__bf16)v.w;
    dst[i] = o;
  }
}

// ---------------------------------------------------------------------------
// t1[m][i] = dot(x[m,:], x[b, 2+2i, :])   (odd tokens of first LoRA half)
// ---------------------------------------------------------------------------
__global__ __launch_bounds__(256) void lora_t1_kernel(
    const float* __restrict__ x, float* __restrict__ t1) {
  const int m = blockIdx.x;
  const int tid = threadIdx.x;
  const int b = m >> kLogS;
  const float* __restrict__ xr = x + (size_t)m * kD;
  const float* __restrict__ od = x + ((size_t)b * kS + 2) * kD;  // tokens 2,4,6,8
  float p[4] = {0.f, 0.f, 0.f, 0.f};
  for (int d = tid; d < kD; d += 256) {
    const float xv = xr[d];
    p[0] += xv * od[0 * kD + d];
    p[1] += xv * od[2 * kD + d];
    p[2] += xv * od[4 * kD + d];
    p[3] += xv * od[6 * kD + d];
  }
  __shared__ float red[4][256];
  #pragma unroll
  for (int i = 0; i < 4; ++i) red[i][tid] = p[i];
  __syncthreads();
  for (int s = 128; s > 0; s >>= 1) {
    if (tid < s) {
      #pragma unroll
      for (int i = 0; i < 4; ++i) red[i][tid] += red[i][tid + s];
    }
    __syncthreads();
  }
  if (tid < 4) t1[m * 4 + tid] = red[tid][0];
}

// ---------------------------------------------------------------------------
// t2[m][i][q] = dot(h[m, q*1024 : q*1024+1024], x[b, 10+2i, :])
// ---------------------------------------------------------------------------
__global__ __launch_bounds__(256) void lora_t2_kernel(
    const float* __restrict__ x, const __bf16* __restrict__ hbf,
    float* __restrict__ t2) {
  const int m = blockIdx.x;
  const int tid = threadIdx.x;
  const int b = m >> kLogS;
  const __bf16* __restrict__ hr = hbf + (size_t)m * kH;
  const float* __restrict__ od = x + ((size_t)b * kS + 10) * kD;  // tokens 10,12,14,16
  float p[16];
  #pragma unroll
  for (int j = 0; j < 16; ++j) p[j] = 0.f;
  for (int c = tid; c < kD; c += 256) {
    float hq[4], ov[4];
    #pragma unroll
    for (int q = 0; q < 4; ++q) hq[q] = (float)hr[q * kD + c];
    #pragma unroll
    for (int i = 0; i < 4; ++i) ov[i] = od[(size_t)(2 * i) * kD + c];
    #pragma unroll
    for (int i = 0; i < 4; ++i)
      #pragma unroll
      for (int q = 0; q < 4; ++q) p[i * 4 + q] += ov[i] * hq[q];
  }
  __shared__ float red[16][256];
  #pragma unroll
  for (int j = 0; j < 16; ++j) red[j][tid] = p[j];
  __syncthreads();
  for (int s = 128; s > 0; s >>= 1) {
    if (tid < s) {
      #pragma unroll
      for (int j = 0; j < 16; ++j) red[j][tid] += red[j][tid + s];
    }
    __syncthreads();
  }
  if (tid < 16) t2[m * 16 + tid] = red[tid][0];
}

// ---------------------------------------------------------------------------
// WMMA fragment gathers from padded LDS tiles (row stride = 40 bf16 = 80 B)
// A 16x32 bf16: lanes 0-15 hold K {0..7,16..23}; lanes 16-31 K {8..15,24..31}
// B 32x16 bf16: lanes 0-15 hold K {0..7,16..23}; lanes 16-31 K {8..15,24..31}
// ---------------------------------------------------------------------------
__device__ __forceinline__ v16bf load_a_frag(const __bf16* p) {
  const v8bf lo = *(const v8bf*)p;          // k = sel .. sel+7
  const v8bf hi = *(const v8bf*)(p + 16);   // k = sel+16 .. sel+23
  v16bf r;
  #pragma unroll
  for (int e = 0; e < 8; ++e) { r[e] = lo[e]; r[e + 8] = hi[e]; }
  return r;
}
__device__ __forceinline__ v16bf load_b_frag(const __bf16* p) {
  const v8bf lo = *(const v8bf*)p;          // k = sel .. sel+7
  const v8bf hi = *(const v8bf*)(p + 16);   // k = sel+16 .. sel+23
  v16bf r;
  #pragma unroll
  for (int e = 0; e < 8; ++e) { r[e] = lo[e]; r[e + 8] = hi[e]; }
  return r;
}

// ---------------------------------------------------------------------------
// GEMM1: h = gelu(xbf @ W1bf^T + scale * lora1 + b1), stored bf16.
// Block tile 256x128, 8 waves (4Mx2N), wave tile 64x64 (4x4 WMMA tiles).
// Software pipelined: register-staged global loads + double-buffered LDS.
// ---------------------------------------------------------------------------
__global__ __launch_bounds__(256) void gemm1_kernel(
    const __bf16* __restrict__ xbf, const __bf16* __restrict__ w1bf,
    const float* __restrict__ x, const float* __restrict__ b1,
    const float* __restrict__ t1, const float* __restrict__ scale_p,
    __bf16* __restrict__ hbf) {
  __shared__ __bf16 ldsA[2][kTileA];
  __shared__ __bf16 ldsB[2][kTileB];
  const int tid = threadIdx.x;
  const int lane = tid & 31;
  const int wv = tid >> 5;
  const int wm = wv & 3;        // 64-row strip (4 strips)
  const int wn = wv >> 2;       // 64-col strip (2 strips)
  const int half = lane >> 4;
  const int lrow = lane & 15;
  const int m0 = blockIdx.x * 256;
  const int n0 = blockIdx.y * 128;

  // per-thread tile coordinates (8 bf16 per chunk; A: 4 chunks, B: 2 chunks)
  const int tRow = tid >> 2, tCol = (tid & 3) * 8;
  const __bf16* __restrict__ gA = &xbf[(size_t)(m0 + tRow) * kD + tCol];
  const __bf16* __restrict__ gB = &w1bf[(size_t)(n0 + tRow) * kD + tCol];
  const int tLds = tRow * kLdsStride + tCol;

  v8f acc[4][4];
  #pragma unroll
  for (int mi = 0; mi < 4; ++mi)
    #pragma unroll
    for (int ni = 0; ni < 4; ++ni)
      #pragma unroll
      for (int e = 0; e < 8; ++e) acc[mi][ni][e] = 0.f;

  v8bf aReg[4], bReg[2];
  // prologue: fetch tile 0 into registers, park in LDS buffer 0
  #pragma unroll
  for (int j = 0; j < 4; ++j) aReg[j] = *(const v8bf*)(gA + (size_t)(64 * j) * kD);
  #pragma unroll
  for (int j = 0; j < 2; ++j) bReg[j] = *(const v8bf*)(gB + (size_t)(64 * j) * kD);
  #pragma unroll
  for (int j = 0; j < 4; ++j) *(v8bf*)&ldsA[0][tLds + 64 * j * kLdsStride] = aReg[j];
  #pragma unroll
  for (int j = 0; j < 2; ++j) *(v8bf*)&ldsB[0][tLds + 64 * j * kLdsStride] = bReg[j];

  int cur = 0;
  for (int k0 = 0; k0 < kD; k0 += 32) {
    const bool hasNext = (k0 + 32) < kD;
    if (hasNext) {   // issue next tile's loads; they overlap the WMMAs below
      #pragma unroll
      for (int j = 0; j < 4; ++j)
        aReg[j] = *(const v8bf*)(gA + (size_t)(64 * j) * kD + (k0 + 32));
      #pragma unroll
      for (int j = 0; j < 2; ++j)
        bReg[j] = *(const v8bf*)(gB + (size_t)(64 * j) * kD + (k0 + 32));
    }
    __syncthreads();   // cur buffer fully written; other buffer fully consumed

    v16bf af[4], bfr[4];
    #pragma unroll
    for (int mi = 0; mi < 4; ++mi)
      af[mi] = load_a_frag(&ldsA[cur][(wm * 64 + mi * 16 + lrow) * kLdsStride + half * 8]);
    #pragma unroll
    for (int ni = 0; ni < 4; ++ni)
      bfr[ni] = load_b_frag(&ldsB[cur][(wn * 64 + ni * 16 + lrow) * kLdsStride + half * 8]);
    #pragma unroll
    for (int mi = 0; mi < 4; ++mi)
      #pragma unroll
      for (int ni = 0; ni < 4; ++ni)
        acc[mi][ni] = __builtin_amdgcn_wmma_f32_16x16x32_bf16(
            false, af[mi], false, bfr[ni], (short)0, acc[mi][ni], false, false);

    if (hasNext) {   // drain staged registers (single loadcnt wait, post-compute)
      const int nxt = cur ^ 1;
      #pragma unroll
      for (int j = 0; j < 4; ++j) *(v8bf*)&ldsA[nxt][tLds + 64 * j * kLdsStride] = aReg[j];
      #pragma unroll
      for (int j = 0; j < 2; ++j) *(v8bf*)&ldsB[nxt][tLds + 64 * j * kLdsStride] = bReg[j];
    }
    cur ^= 1;
  }

  // Epilogue: + b1 + scale*t1[m,i]*even[b,i,r], exact GELU, bf16 store.
  const float sc = *scale_p;
  const int bi = m0 >> kLogS;            // 256 rows never cross a batch
  const int i1 = n0 >> 10;               // 128 cols never cross a 1024 block
  const float* __restrict__ ev = &x[((size_t)bi * kS + 1 + 2 * i1) * kD];
  #pragma unroll
  for (int ni = 0; ni < 4; ++ni) {
    const int n = n0 + wn * 64 + ni * 16 + lrow;
    const float bias = b1[n];
    const float evv = ev[n & 1023];
    #pragma unroll
    for (int mi = 0; mi < 4; ++mi) {
      #pragma unroll
      for (int e = 0; e < 8; ++e) {
        const int m = m0 + wm * 64 + mi * 16 + half * 8 + e;
        float v = acc[mi][ni][e] + bias + sc * t1[m * 4 + i1] * evv;
        v = 0.5f * v * (1.0f + erff(v * 0.70710678118654752440f));
        hbf[(size_t)m * kH + n] = (__bf16)v;
      }
    }
  }
}

// ---------------------------------------------------------------------------
// GEMM2: out = h @ W2^T + scale * lora2 + b2, fp32 store. Same pipeline.
// ---------------------------------------------------------------------------
__global__ __launch_bounds__(256) void gemm2_kernel(
    const __bf16* __restrict__ hbf, const __bf16* __restrict__ w2bf,
    const float* __restrict__ b2, const float* __restrict__ t2,
    const float* __restrict__ x, const float* __restrict__ scale_p,
    float* __restrict__ out) {
  __shared__ __bf16 ldsA[2][kTileA];
  __shared__ __bf16 ldsB[2][kTileB];
  const int tid = threadIdx.x;
  const int lane = tid & 31;
  const int wv = tid >> 5;
  const int wm = wv & 3;
  const int wn = wv >> 2;
  const int half = lane >> 4;
  const int lrow = lane & 15;
  const int m0 = blockIdx.x * 256;
  const int n0 = blockIdx.y * 128;

  const int tRow = tid >> 2, tCol = (tid & 3) * 8;
  const __bf16* __restrict__ gA = &hbf[(size_t)(m0 + tRow) * kH + tCol];
  const __bf16* __restrict__ gB = &w2bf[(size_t)(n0 + tRow) * kH + tCol];
  const int tLds = tRow * kLdsStride + tCol;

  v8f acc[4][4];
  #pragma unroll
  for (int mi = 0; mi < 4; ++mi)
    #pragma unroll
    for (int ni = 0; ni < 4; ++ni)
      #pragma unroll
      for (int e = 0; e < 8; ++e) acc[mi][ni][e] = 0.f;

  v8bf aReg[4], bReg[2];
  #pragma unroll
  for (int j = 0; j < 4; ++j) aReg[j] = *(const v8bf*)(gA + (size_t)(64 * j) * kH);
  #pragma unroll
  for (int j = 0; j < 2; ++j) bReg[j] = *(const v8bf*)(gB + (size_t)(64 * j) * kH);
  #pragma unroll
  for (int j = 0; j < 4; ++j) *(v8bf*)&ldsA[0][tLds + 64 * j * kLdsStride] = aReg[j];
  #pragma unroll
  for (int j = 0; j < 2; ++j) *(v8bf*)&ldsB[0][tLds + 64 * j * kLdsStride] = bReg[j];

  int cur = 0;
  for (int k0 = 0; k0 < kH; k0 += 32) {
    const bool hasNext = (k0 + 32) < kH;
    if (hasNext) {
      #pragma unroll
      for (int j = 0; j < 4; ++j)
        aReg[j] = *(const v8bf*)(gA + (size_t)(64 * j) * kH + (k0 + 32));
      #pragma unroll
      for (int j = 0; j < 2; ++j)
        bReg[j] = *(const v8bf*)(gB + (size_t)(64 * j) * kH + (k0 + 32));
    }
    __syncthreads();

    v16bf af[4], bfr[4];
    #pragma unroll
    for (int mi = 0; mi < 4; ++mi)
      af[mi] = load_a_frag(&ldsA[cur][(wm * 64 + mi * 16 + lrow) * kLdsStride + half * 8]);
    #pragma unroll
    for (int ni = 0; ni < 4; ++ni)
      bfr[ni] = load_b_frag(&ldsB[cur][(wn * 64 + ni * 16 + lrow) * kLdsStride + half * 8]);
    #pragma unroll
    for (int mi = 0; mi < 4; ++mi)
      #pragma unroll
      for (int ni = 0; ni < 4; ++ni)
        acc[mi][ni] = __builtin_amdgcn_wmma_f32_16x16x32_bf16(
            false, af[mi], false, bfr[ni], (short)0, acc[mi][ni], false, false);

    if (hasNext) {
      const int nxt = cur ^ 1;
      #pragma unroll
      for (int j = 0; j < 4; ++j) *(v8bf*)&ldsA[nxt][tLds + 64 * j * kLdsStride] = aReg[j];
      #pragma unroll
      for (int j = 0; j < 2; ++j) *(v8bf*)&ldsB[nxt][tLds + 64 * j * kLdsStride] = bReg[j];
    }
    cur ^= 1;
  }

  // Epilogue: + b2 + scale * sum_q even2[b,i,(o&255)*4+q] * t2[m,i,q]
  const float sc = *scale_p;
  const int bi = m0 >> kLogS;
  const int i2 = n0 >> 8;                // 128 cols never cross a 256 block
  const float* __restrict__ ev = &x[((size_t)bi * kS + 9 + 2 * i2) * kD];
  #pragma unroll
  for (int ni = 0; ni < 4; ++ni) {
    const int o = n0 + wn * 64 + ni * 16 + lrow;
    const float bias = b2[o];
    const int r0 = (o & 255) * 4;
    const float s0 = sc * ev[r0 + 0], s1 = sc * ev[r0 + 1];
    const float s2 = sc * ev[r0 + 2], s3 = sc * ev[r0 + 3];
    #pragma unroll
    for (int mi = 0; mi < 4; ++mi) {
      #pragma unroll
      for (int e = 0; e < 8; ++e) {
        const int m = m0 + wm * 64 + mi * 16 + half * 8 + e;
        const float4 t = *(const float4*)&t2[m * 16 + i2 * 4];
        out[(size_t)m * kD + o] =
            acc[mi][ni][e] + bias + s0 * t.x + s1 * t.y + s2 * t.z + s3 * t.w;
      }
    }
  }
}

// ---------------------------------------------------------------------------
extern "C" void kernel_launch(void* const* d_in, const int* in_sizes, int n_in,
                              void* d_out, int out_size, void* d_ws, size_t ws_size,
                              hipStream_t stream) {
  (void)in_sizes; (void)n_in; (void)out_size; (void)ws_size;
  const float* x     = (const float*)d_in[0];
  const float* W1    = (const float*)d_in[1];
  const float* b1    = (const float*)d_in[2];
  const float* W2    = (const float*)d_in[3];
  const float* b2    = (const float*)d_in[4];
  const float* scale = (const float*)d_in[5];
  float* out = (float*)d_out;

  char* ws = (char*)d_ws;
  __bf16* W1bf = (__bf16*)(ws + OFF_W1BF);
  __bf16* W2bf = (__bf16*)(ws + OFF_W2BF);
  __bf16* xbf  = (__bf16*)(ws + OFF_XBF);
  __bf16* hbf  = (__bf16*)(ws + OFF_HBF);
  float*  t1   = (float*)(ws + OFF_T1);
  float*  t2   = (float*)(ws + OFF_T2);

  const int n4w = kH * kD / 4;   // per weight matrix
  const int n4x = kM * kD / 4;   // activations
  cvt_f32_to_bf16_x4<<<2048, 256, 0, stream>>>((const float4*)W1, (v4bf*)W1bf, n4w);
  cvt_f32_to_bf16_x4<<<2048, 256, 0, stream>>>((const float4*)W2, (v4bf*)W2bf, n4w);
  cvt_f32_to_bf16_x4<<<4096, 256, 0, stream>>>((const float4*)x,  (v4bf*)xbf,  n4x);

  lora_t1_kernel<<<kM, 256, 0, stream>>>(x, t1);

  gemm1_kernel<<<dim3(kM / 256, kH / 128), 256, 0, stream>>>(xbf, W1bf, x, b1, t1, scale, hbf);

  lora_t2_kernel<<<kM, 256, 0, stream>>>(x, hbf, t2);

  gemm2_kernel<<<dim3(kM / 256, kD / 128), 256, 0, stream>>>(hbf, W2bf, b2, t2, x, scale, out);
}
